// MixtralSparseMoeBlock_49486613184823
// MI455X (gfx1250) — matrix operand order repaired and checked by reading, not compiled, not run on hardware
//
#include <hip/hip_runtime.h>
#include <hip/hip_bf16.h>

// ---------------- WMMA fragment types (gfx1250, wave32) ----------------
typedef __bf16 v16bf __attribute__((ext_vector_type(16)));
typedef __bf16 v8bf  __attribute__((ext_vector_type(8)));
typedef float  v8f   __attribute__((ext_vector_type(8)));
// ------- TDM descriptor groups (6-arg clang-23 builtin signature) ------
typedef unsigned int u32x4 __attribute__((ext_vector_type(4)));
typedef int          i32x8 __attribute__((ext_vector_type(8)));
typedef int          i32x4 __attribute__((ext_vector_type(4)));

#define T_TOK 65536
#define DDIM  1024
#define PDIM  512
#define BM 128
#define BN 128
#define BK 32

// =======================================================================
// GEMM1: patches = tanh(x @ W_t1), fused bag column-sum
//   A tile (128x32 f32) staged by the Tensor Data Mover (TENSORcnt),
//   B tile staged transposed bf16 with packed b32 DS stores.
// 256 threads = 8 waves (2 M x 4 N), wave tile 64x32 = 4x2 WMMA 16x16x32
// =======================================================================
__global__ __launch_bounds__(256) void k_gemm1(
    const float* __restrict__ x, const float* __restrict__ Wt1,
    float* __restrict__ patches, float* __restrict__ bagsum)
{
  __shared__ float  sAf[BM * BK];      // f32 A tile, written by TDM
  __shared__ __bf16 sB[BN * BK];       // transposed (n, k) bf16
  __shared__ float  scol[BN];

  const int m0 = blockIdx.x * BM;
  const int n0 = blockIdx.y * BN;
  const int tid  = threadIdx.x;
  const int wave = tid >> 5, lane = tid & 31;
  const int wm = wave >> 2, wn = wave & 3;          // 2 x 4 wave grid
  const int laneM = lane & 15, laneH = lane >> 4;

  v8f acc[4][2] = {};

  const int bk2 = (tid >> 4) * 2;            // even k within chunk: 0..30
  const int bn8 = (tid & 15) * 8;            // 0..120

  for (int k0 = 0; k0 < DDIM; k0 += BK) {
    // ---- B staging: 2 k-rows x 8 n per thread, packed bf16x2 stores ----
    {
      const float* s0 = Wt1 + (size_t)(k0 + bk2) * PDIM + n0 + bn8;
      const float* s1 = s0 + PDIM;
      float4 a0 = *(const float4*)(s0);
      float4 a1 = *(const float4*)(s0 + 4);
      float4 b0 = *(const float4*)(s1);
      float4 b1 = *(const float4*)(s1 + 4);
      float va[8] = {a0.x, a0.y, a0.z, a0.w, a1.x, a1.y, a1.z, a1.w};
      float vb[8] = {b0.x, b0.y, b0.z, b0.w, b1.x, b1.y, b1.z, b1.w};
      #pragma unroll
      for (int i = 0; i < 8; i++) {
        unsigned short lo = __builtin_bit_cast(unsigned short, (__bf16)va[i]);
        unsigned short hi = __builtin_bit_cast(unsigned short, (__bf16)vb[i]);
        *(unsigned*)&sB[(bn8 + i) * BK + bk2] = (unsigned)lo | ((unsigned)hi << 16);
      }
    }
    // ---- A staging via Tensor Data Mover: 128 rows x 32 f32, wave 0 ----
    if (tid < 32) {
      unsigned long long ga =
          (unsigned long long)(uintptr_t)(x + (size_t)m0 * DDIM + k0);
      unsigned lds = (unsigned)(uintptr_t)(void*)sAf;   // LDS byte address
      u32x4 g0;
      g0[0] = 1u;                                       // count=1, user D#
      g0[1] = lds;                                      // lds_addr
      g0[2] = (unsigned)(ga & 0xFFFFFFFFu);             // global_addr lo
      g0[3] = (unsigned)((ga >> 32) & 0x01FFFFFFu)      // global_addr hi
              | (2u << 30);                             // type = 2 (image)
      i32x8 g1;
      g1[0] = (int)(2u << 16);                          // data_size=4B
      g1[1] = (int)((DDIM & 0xFFFF) << 16);             // tensor_dim0 lo16
      g1[2] = (int)((DDIM >> 16) | ((T_TOK & 0xFFFF) << 16)); // d0 hi | d1 lo
      g1[3] = (int)(((T_TOK >> 16) & 0xFFFF) | (BK << 16));   // d1 hi | tile0
      g1[4] = (int)BM;                                  // tile_dim1, tile2=0
      g1[5] = (int)DDIM;                                // dim0_stride lo32
      g1[6] = 0;                                        // stride hi / d1 stride
      g1[7] = 0;
      i32x4 gz4 = {0, 0, 0, 0};                         // 2D: groups 2/3 unused
      i32x8 gz8 = {0, 0, 0, 0, 0, 0, 0, 0};
      __builtin_amdgcn_tensor_load_to_lds(g0, g1, gz4, gz4, gz8, 0);
      __builtin_amdgcn_s_wait_tensorcnt((short)0);
    }
    __syncthreads();

    if (k0 + BK < DDIM)  // L2 prefetch of next W chunk -> global_prefetch_b8
      __builtin_prefetch(Wt1 + (size_t)(k0 + BK + bk2) * PDIM + n0 + bn8, 0, 1);

    // ---- B fragments: 16 contiguous bf16 per lane ----
    v16bf bf[2];
    #pragma unroll
    for (int ni = 0; ni < 2; ni++) {
      int col = wn * 32 + ni * 16 + laneM;
      bf[ni] = *(const v16bf*)&sB[col * BK + laneH * 16];
    }
    // ---- A fragments from f32 LDS tile, cvt to bf16 (VALU co-exec) ----
    #pragma unroll
    for (int mi = 0; mi < 4; mi++) {
      int row = wm * 64 + mi * 16 + laneM;
      const float* ap = &sAf[row * BK + laneH * 16];
      v16bf a;
      #pragma unroll
      for (int i = 0; i < 8; i++) a[i] = (__bf16)ap[i];
      #pragma unroll
      for (int i = 0; i < 8; i++) a[8 + i] = (__bf16)ap[8 + i];
      #pragma unroll
      for (int ni = 0; ni < 2; ni++)
        acc[mi][ni] = __builtin_amdgcn_wmma_f32_16x16x32_bf16(
            false, a, false, bf[ni], (short)0, acc[mi][ni], false, false);
    }
    __syncthreads();
  }

  // ---- epilogue: tanh, store patches, column-sum for bag ----
  if (tid < BN) scol[tid] = 0.f;
  __syncthreads();
  #pragma unroll
  for (int ni = 0; ni < 2; ni++) {
    int col = n0 + wn * 32 + ni * 16 + laneM;
    float cp = 0.f;
    #pragma unroll
    for (int mi = 0; mi < 4; mi++) {
      #pragma unroll
      for (int r = 0; r < 8; r++) {
        int row = m0 + wm * 64 + mi * 16 + laneH * 8 + r;
        float v = tanhf(acc[mi][ni][r]);
        patches[(size_t)row * PDIM + col] = v;
        cp += v;
      }
    }
    atomicAdd(&scol[wn * 32 + ni * 16 + laneM], cp);   // ds_add_f32
  }
  __syncthreads();
  if (tid < BN) atomicAdd(&bagsum[n0 + tid], scol[tid]);
}

// =======================================================================
// Router: 3 gates, softmax/top-1, sortable keys, mask bits, counts.
// =======================================================================
__global__ __launch_bounds__(256) void k_router(
    const float* __restrict__ patches,
    const float* __restrict__ Wg0, const float* __restrict__ Wg1,
    const float* __restrict__ Wg2,
    float* __restrict__ out_router, unsigned* __restrict__ keys,
    unsigned char* __restrict__ selbits, int* __restrict__ cnt)
{
  __shared__ float g[3][PDIM * 2];
  int tid = threadIdx.x;
  for (int i = tid; i < PDIM * 2; i += 256) {
    g[0][i] = Wg0[i]; g[1][i] = Wg1[i]; g[2][i] = Wg2[i];
  }
  __syncthreads();
  int wave = tid >> 5, lane = tid & 31;
  int gw = blockIdx.x * 8 + wave;
  int nwaves = gridDim.x * 8;
  int c0 = 0, c1 = 0, c2 = 0;
  for (int t = gw; t < T_TOK; t += nwaves) {
    const float* row = patches + (size_t)t * PDIM + lane * 16;
    float s[6] = {0, 0, 0, 0, 0, 0};
    #pragma unroll
    for (int i = 0; i < 16; i += 4) {
      float4 v = *(const float4*)(row + i);
      float vv[4] = {v.x, v.y, v.z, v.w};
      #pragma unroll
      for (int j = 0; j < 4; j++) {
        int idx = (lane * 16 + i + j) * 2;
        float pv = vv[j];
        s[0] += pv * g[0][idx]; s[1] += pv * g[0][idx + 1];
        s[2] += pv * g[1][idx]; s[3] += pv * g[1][idx + 1];
        s[4] += pv * g[2][idx]; s[5] += pv * g[2][idx + 1];
      }
    }
    #pragma unroll
    for (int off = 16; off; off >>= 1)
      #pragma unroll
      for (int j = 0; j < 6; j++) s[j] += __shfl_xor(s[j], off, 32);
    if (lane == 0) {
      out_router[(size_t)t * 2 + 0] = s[0];
      out_router[(size_t)t * 2 + 1] = s[1];
      float m = fmaxf(s[0], s[1]);
      float e0 = __expf(s[0] - m), e1 = __expf(s[1] - m);
      float wtop = fmaxf(e0, e1) / (e0 + e1);
      unsigned u = __float_as_uint(wtop);
      u = (u >> 31) ? ~u : (u | 0x80000000u);   // sortable-uint map
      keys[t] = u;
      int b0 = s[1] > s[0], b1 = s[3] > s[2], b2 = s[5] > s[4];
      selbits[t] = (unsigned char)(b0 | (b1 << 1) | (b2 << 2));
      c0 += b0; c1 += b1; c2 += b2;
    }
  }
  if (lane == 0) {
    if (c0) atomicAdd(&cnt[0], c0);
    if (c1) atomicAdd(&cnt[1], c1);
    if (c2) atomicAdd(&cnt[2], c2);
  }
}

// ----------------------- radix-select control ---------------------------
__global__ void k_init_sel(const int* __restrict__ cnt, int* nums, int* remain,
                           unsigned* prefix, float* dist_out)
{
  int e = threadIdx.x;
  if (e < 3) {
    const float fr[3] = {0.25f, 0.5f, 0.25f};
    int c = cnt[e];
    int n = (int)floorf((float)c * fr[e]);
    if (n == 0) n = c;
    nums[e] = n; remain[e] = n; prefix[e] = 0u;
    dist_out[1 + e] = (float)n;
  }
  if (e == 0) dist_out[0] = (float)T_TOK;
}

__global__ __launch_bounds__(256) void k_hist(
    const unsigned* __restrict__ keys, const unsigned char* __restrict__ selbits,
    const unsigned* __restrict__ prefix, unsigned* __restrict__ hist, int pass)
{
  __shared__ unsigned lh[3 * 256];
  int tid = threadIdx.x;
  for (int i = tid; i < 768; i += 256) lh[i] = 0u;
  __syncthreads();
  unsigned pf0 = prefix[0], pf1 = prefix[1], pf2 = prefix[2];
  int shift = 24 - 8 * pass;
  for (int t = blockIdx.x * 256 + tid; t < T_TOK; t += gridDim.x * 256) {
    unsigned k = keys[t];
    unsigned char sb = selbits[t];
    unsigned b = (k >> shift) & 255u;
    if ((sb & 1) && (pass == 0 || (k >> (shift + 8)) == (pf0 >> (shift + 8))))
      atomicAdd(&lh[0 * 256 + b], 1u);
    if ((sb & 2) && (pass == 0 || (k >> (shift + 8)) == (pf1 >> (shift + 8))))
      atomicAdd(&lh[1 * 256 + b], 1u);
    if ((sb & 4) && (pass == 0 || (k >> (shift + 8)) == (pf2 >> (shift + 8))))
      atomicAdd(&lh[2 * 256 + b], 1u);
  }
  __syncthreads();
  for (int i = tid; i < 768; i += 256)
    if (lh[i]) atomicAdd(&hist[i], lh[i]);
}

__global__ void k_scan(const unsigned* __restrict__ hist, unsigned* prefix,
                       int* remain, int* need, int pass)
{
  int e = threadIdx.x;
  if (e >= 3) return;
  int shift = 24 - 8 * pass;
  int rem = remain[e];
  unsigned pf = prefix[e];
  for (int b = 255; b >= 0; b--) {
    int c = (int)hist[e * 256 + b];
    if (rem > c) { rem -= c; }
    else { pf |= ((unsigned)b) << shift; break; }
  }
  prefix[e] = pf; remain[e] = rem;
  if (pass == 3) need[e] = rem;
}

__global__ __launch_bounds__(256) void k_select(
    const unsigned* __restrict__ keys, const unsigned char* __restrict__ selbits,
    const unsigned* __restrict__ prefix, const int* __restrict__ need,
    unsigned* tie, unsigned char* __restrict__ finalsel)
{
  unsigned pf0 = prefix[0], pf1 = prefix[1], pf2 = prefix[2];
  int nd0 = need[0], nd1 = need[1], nd2 = need[2];
  for (int t = blockIdx.x * 256 + threadIdx.x; t < T_TOK; t += gridDim.x * 256) {
    unsigned k = keys[t];
    unsigned char sb = selbits[t], f = 0;
    if (sb & 1) {
      if (k > pf0) f |= 1;
      else if (k == pf0) { if ((int)atomicAdd(&tie[0], 1u) < nd0) f |= 1; }
    }
    if (sb & 2) {
      if (k > pf1) f |= 2;
      else if (k == pf1) { if ((int)atomicAdd(&tie[1], 1u) < nd1) f |= 2; }
    }
    if (sb & 4) {
      if (k > pf2) f |= 4;
      else if (k == pf2) { if ((int)atomicAdd(&tie[2], 1u) < nd2) f |= 4; }
    }
    finalsel[t] = f;
  }
}

// ------------------- per-expert masked feature sums ---------------------
__global__ __launch_bounds__(256) void k_meanfeat(
    const float* __restrict__ patches, const unsigned char* __restrict__ finalsel,
    float* __restrict__ macc)
{
  __shared__ float acc[3 * PDIM];
  int tid = threadIdx.x;
  for (int i = tid; i < 3 * PDIM; i += 256) acc[i] = 0.f;
  __syncthreads();
  int t0 = blockIdx.x * 64;
  for (int tt = 0; tt < 64; tt++) {
    int t = t0 + tt;
    unsigned char f = finalsel[t];
    if (!f) continue;
    const float* row = patches + (size_t)t * PDIM;
    for (int p = tid; p < PDIM; p += 256) {
      float v = row[p];
      if (f & 1) acc[p] += v;
      if (f & 2) acc[PDIM + p] += v;
      if (f & 4) acc[2 * PDIM + p] += v;
    }
  }
  __syncthreads();
  for (int i = tid; i < 3 * PDIM; i += 256) {
    float v = acc[i];
    if (v != 0.f) atomicAdd(&macc[i], v);
  }
}

// ----------------------------- finalize 1 -------------------------------
__device__ inline float block_reduce_sum(float v, float* red)
{
  int tid = threadIdx.x;
  red[tid] = v; __syncthreads();
  for (int s = 128; s > 0; s >>= 1) {
    if (tid < s) red[tid] += red[tid + s];
    __syncthreads();
  }
  float r = red[0]; __syncthreads();
  return r;
}

__global__ __launch_bounds__(256) void k_final1(
    const float* __restrict__ bagsum, const float* __restrict__ Wcls1,
    const float* __restrict__ macc, const int* __restrict__ nums,
    const float* __restrict__ Wclf, float* __restrict__ out)
{
  __shared__ float red[256];
  int tid = threadIdx.x;
  float p0 = 0.f, p1 = 0.f;
  for (int p = tid; p < PDIM; p += 256) {
    float b = bagsum[p] * (1.f / (float)T_TOK);
    p0 += b * Wcls1[p * 2 + 0];
    p1 += b * Wcls1[p * 2 + 1];
  }
  float yl0 = block_reduce_sum(p0, red);
  float yl1 = block_reduce_sum(p1, red);
  float lg[3][3];
  for (int e = 0; e < 3; e++) {
    float inv = 1.f / (float)max(nums[e], 1);
    for (int j = 0; j < 3; j++) {
      float s = 0.f;
      for (int p = tid; p < PDIM; p += 256)
        s += macc[e * PDIM + p] * inv * Wclf[(size_t)(e * PDIM + p) * 3 + j];
      lg[e][j] = block_reduce_sum(s, red);
    }
  }
  if (tid == 0) {
    out[3] = yl0; out[4] = yl1;
    float m = fmaxf(yl0, yl1);
    float e0 = __expf(yl0 - m), e1 = __expf(yl1 - m), den = e0 + e1;
    out[5] = e0 / den; out[6] = e1 / den;
    out[7] = (yl1 > yl0) ? 1.f : 0.f;
    float joint = 0.f;
    for (int e = 0; e < 2; e++) {
      float mm = fmaxf(fmaxf(lg[e][0], lg[e][1]), lg[e][2]);
      float lse = logf(__expf(lg[e][0] - mm) + __expf(lg[e][1] - mm) +
                       __expf(lg[e][2] - mm));
      joint += -(lg[e][e] - mm - lse);
    }
    out[131080] = joint;
  }
}

// =======================================================================
// GEMM2: agg_acc[n] += sum_t total_sel[t] * tanh(patches[t] @ W_a1)[n]
// Pure bf16-LDS WMMA path, feats2 never materialized.
// =======================================================================
__global__ __launch_bounds__(256) void k_gemm2(
    const float* __restrict__ patches, const float* __restrict__ Wa1,
    const unsigned char* __restrict__ finalsel, float* __restrict__ aggacc)
{
  __shared__ __bf16 sA[BM * BK];
  __shared__ __bf16 sB[BN * BK];
  __shared__ float  scol[BN];
  __shared__ float  sW[BM];

  const int m0 = blockIdx.x * BM;
  const int n0 = blockIdx.y * BN;
  const int tid  = threadIdx.x;
  const int wave = tid >> 5, lane = tid & 31;
  const int wm = wave >> 2, wn = wave & 3;
  const int laneM = lane & 15, laneH = lane >> 4;

  if (tid < BM) sW[tid] = (float)__popc((int)(finalsel[m0 + tid] & 7));

  v8f acc[4][2] = {};
  const int arow = tid >> 1;
  const int ach  = (tid & 1) * 16;
  const int bk2 = (tid >> 4) * 2;
  const int bn8 = (tid & 15) * 8;

  for (int k0 = 0; k0 < PDIM; k0 += BK) {
    {
      const float* src = patches + (size_t)(m0 + arow) * PDIM + k0 + ach;
      #pragma unroll
      for (int i = 0; i < 16; i += 4) {
        float4 v = *(const float4*)(src + i);
        sA[arow * BK + ach + i + 0] = (__bf16)v.x;
        sA[arow * BK + ach + i + 1] = (__bf16)v.y;
        sA[arow * BK + ach + i + 2] = (__bf16)v.z;
        sA[arow * BK + ach + i + 3] = (__bf16)v.w;
      }
    }
    {
      const float* s0 = Wa1 + (size_t)(k0 + bk2) * PDIM + n0 + bn8;
      const float* s1 = s0 + PDIM;
      float4 a0 = *(const float4*)(s0);
      float4 a1 = *(const float4*)(s0 + 4);
      float4 b0 = *(const float4*)(s1);
      float4 b1 = *(const float4*)(s1 + 4);
      float va[8] = {a0.x, a0.y, a0.z, a0.w, a1.x, a1.y, a1.z, a1.w};
      float vb[8] = {b0.x, b0.y, b0.z, b0.w, b1.x, b1.y, b1.z, b1.w};
      #pragma unroll
      for (int i = 0; i < 8; i++) {
        unsigned short lo = __builtin_bit_cast(unsigned short, (__bf16)va[i]);
        unsigned short hi = __builtin_bit_cast(unsigned short, (__bf16)vb[i]);
        *(unsigned*)&sB[(bn8 + i) * BK + bk2] = (unsigned)lo | ((unsigned)hi << 16);
      }
    }
    __syncthreads();
    if (k0 + BK < PDIM)
      __builtin_prefetch(patches + (size_t)(m0 + arow) * PDIM + k0 + BK + ach, 0, 1);

    v16bf bf[2];
    #pragma unroll
    for (int ni = 0; ni < 2; ni++) {
      int col = wn * 32 + ni * 16 + laneM;
      bf[ni] = *(const v16bf*)&sB[col * BK + laneH * 16];
    }
    #pragma unroll
    for (int mi = 0; mi < 4; mi++) {
      int row = wm * 64 + mi * 16 + laneM;
      v8bf lo = *(const v8bf*)&sA[row * BK + laneH * 16];
      v8bf hi = *(const v8bf*)&sA[row * BK + laneH * 16 + 8];
      v16bf a;
      #pragma unroll
      for (int i = 0; i < 8; i++) { a[i] = lo[i]; a[8 + i] = hi[i]; }
      #pragma unroll
      for (int ni = 0; ni < 2; ni++)
        acc[mi][ni] = __builtin_amdgcn_wmma_f32_16x16x32_bf16(
            false, a, false, bf[ni], (short)0, acc[mi][ni], false, false);
    }
    __syncthreads();
  }

  if (tid < BN) scol[tid] = 0.f;
  __syncthreads();
  #pragma unroll
  for (int ni = 0; ni < 2; ni++) {
    float cp = 0.f;
    #pragma unroll
    for (int mi = 0; mi < 4; mi++) {
      #pragma unroll
      for (int r = 0; r < 8; r++) {
        int lr = wm * 64 + mi * 16 + laneH * 8 + r;
        cp += tanhf(acc[mi][ni][r]) * sW[lr];
      }
    }
    atomicAdd(&scol[wn * 32 + ni * 16 + laneM], cp);
  }
  __syncthreads();
  if (tid < BN) atomicAdd(&aggacc[n0 + tid], scol[tid]);
}

// ----------------------------- finalize 2 -------------------------------
__global__ __launch_bounds__(256) void k_final2(
    const float* __restrict__ aggacc, const int* __restrict__ nums,
    const float* __restrict__ Wacls, float* __restrict__ out)
{
  __shared__ float red[256];
  int tid = threadIdx.x;
  float tn = (float)(nums[0] + nums[1] + nums[2]);
  float inv = 1.f / fmaxf(tn, 1.f);
  float p0 = 0.f, p1 = 0.f;
  for (int p = tid; p < PDIM; p += 256) {
    float a = aggacc[p] * inv;
    p0 += a * Wacls[p * 2 + 0];
    p1 += a * Wacls[p * 2 + 1];
  }
  float l0 = block_reduce_sum(p0, red);
  float l1 = block_reduce_sum(p1, red);
  if (tid == 0) {
    out[0] = l0; out[1] = l1;
    out[2] = (l1 > l0) ? 1.f : 0.f;
  }
}

// =======================================================================
extern "C" void kernel_launch(void* const* d_in, const int* in_sizes, int n_in,
                              void* d_out, int out_size, void* d_ws, size_t ws_size,
                              hipStream_t stream)
{
  const float* x     = (const float*)d_in[0];   // (1,65536,1024)
  const float* Wt1   = (const float*)d_in[1];   // (1024,512)
  const float* Wcls1 = (const float*)d_in[2];   // (512,2)
  const float* Wg0   = (const float*)d_in[3];   // (512,2)
  const float* Wg1   = (const float*)d_in[4];
  const float* Wg2   = (const float*)d_in[5];
  const float* Wclf  = (const float*)d_in[6];   // (3,512,3)
  const float* Wa1   = (const float*)d_in[7];   // (512,512)
  const float* Wacls = (const float*)d_in[8];   // (512,2)
  float* out = (float*)d_out;

  char* W = (char*)d_ws;
  size_t o = 0;
  float* patches = (float*)(W + o);          o += (size_t)T_TOK * PDIM * 4;
  unsigned* keys = (unsigned*)(W + o);       o += (size_t)T_TOK * 4;
  unsigned char* selbits  = (unsigned char*)(W + o); o += T_TOK;
  unsigned char* finalsel = (unsigned char*)(W + o); o += T_TOK;
  float* bagsum = (float*)(W + o);
  float* aggacc = bagsum + PDIM;
  float* macc   = aggacc + PDIM;             // 3*512
  int*   cnt    = (int*)(macc + 3 * PDIM);   // 4
  unsigned* tie = (unsigned*)(cnt + 4);      // 4
  size_t zbytes = (size_t)(PDIM + PDIM + 3 * PDIM) * 4 + 4 * 4 + 4 * 4;
  int* nums   = (int*)(tie + 4);
  int* remain = nums + 4;
  int* need   = remain + 4;
  unsigned* prefix = (unsigned*)(need + 4);  // 4
  unsigned* hist   = prefix + 4;             // 3*256

  (void)hipMemsetAsync(bagsum, 0, zbytes, stream);

  k_gemm1<<<dim3(T_TOK / BM, PDIM / BN), 256, 0, stream>>>(x, Wt1, patches, bagsum);

  k_router<<<2048, 256, 0, stream>>>(patches, Wg0, Wg1, Wg2,
                                     out + 8, keys, selbits, cnt);

  k_init_sel<<<1, 32, 0, stream>>>(cnt, nums, remain, prefix, out + 131081);

  for (int pass = 0; pass < 4; pass++) {
    (void)hipMemsetAsync(hist, 0, 3 * 256 * 4, stream);
    k_hist<<<128, 256, 0, stream>>>(keys, selbits, prefix, hist, pass);
    k_scan<<<1, 32, 0, stream>>>(hist, prefix, remain, need, pass);
  }

  k_select<<<256, 256, 0, stream>>>(keys, selbits, prefix, need, tie, finalsel);

  k_meanfeat<<<T_TOK / 64, 256, 0, stream>>>(patches, finalsel, macc);

  k_final1<<<1, 256, 0, stream>>>(bagsum, Wcls1, macc, nums, Wclf, out);

  k_gemm2<<<dim3(T_TOK / BM, PDIM / BN), 256, 0, stream>>>(patches, Wa1, finalsel, aggacc);

  k_final2<<<1, 256, 0, stream>>>(aggacc, nums, Wacls, out);
}
